// MultiHeadedAttention_42674795053719
// MI455X (gfx1250) — matrix operand order repaired and checked by reading, not compiled, not run on hardware
//
#include <hip/hip_runtime.h>
#include <hip/hip_bf16.h>
#include <math.h>

#define B_  16
#define N_  128
#define D_  256
#define H_  8
#define DK_ 32

typedef __attribute__((ext_vector_type(16))) _Float16 v16h;
typedef __attribute__((ext_vector_type(8)))  float    v8f;
typedef __attribute__((ext_vector_type(4)))  unsigned int v4u;
typedef __attribute__((ext_vector_type(8)))  int v8i_;
typedef __attribute__((ext_vector_type(4)))  int v4i_;

union V16 { v16h v; uint4 q[2]; };
union PK8 { _Float16 h[8]; uint4 q; };
union PK4 { _Float16 h[4]; uint2 u; };

#if __has_builtin(__builtin_amdgcn_tensor_load_to_lds)
#define HAVE_TDM 1
#else
#define HAVE_TDM 0
#endif

// ---------------------------------------------------------------------------
// Kernel A: qs = (query@Wq + bq) * (1/sqrt(DK)),  vlin = value@Wv + bv
// ---------------------------------------------------------------------------
__global__ __launch_bounds__(256) void gt_qv_kernel(
    const float* __restrict__ query, const float* __restrict__ value,
    const float* __restrict__ Wq, const float* __restrict__ bq,
    const float* __restrict__ Wv, const float* __restrict__ bv,
    float* __restrict__ qs, float* __restrict__ vlin)
{
    const int idx = blockIdx.x * 256 + threadIdx.x;
    const int d  = idx & (D_ - 1);
    const int bn = idx >> 8;
    const float* qrow = query + (size_t)bn * D_;
    const float* vrow = value + (size_t)bn * D_;
    float accq = bq[d];
    float accv = bv[d];
    for (int k = 0; k < D_; ++k) {
        accq = fmaf(qrow[k], Wq[k * D_ + d], accq);
        accv = fmaf(vrow[k], Wv[k * D_ + d], accv);
    }
    qs[idx]   = accq * 0.17677669529663687f;   // 1/sqrt(32) folded into q
    vlin[idx] = accv;
}

// ---------------------------------------------------------------------------
// Kernel B: pack Wk (fp32 row-major [k][n]) -> f16 transposed [n][k]
// ---------------------------------------------------------------------------
__global__ __launch_bounds__(256) void gt_pack_wk_kernel(
    const float* __restrict__ Wk, _Float16* __restrict__ wkT)
{
    const int idx = blockIdx.x * 256 + threadIdx.x;  // 65536
    const int n = idx >> 8, k = idx & (D_ - 1);
    wkT[n * D_ + k] = (_Float16)Wk[k * D_ + n];
}

// ---------------------------------------------------------------------------
// Kernel C: adjsm[b,i,:] = softmax_j( mask? -lam*adj[b,i,j] : -inf )
// ---------------------------------------------------------------------------
__global__ __launch_bounds__(32) void gt_adj_softmax_kernel(
    const float* __restrict__ adj, const int* __restrict__ mask,
    const float* __restrict__ lam, float* __restrict__ adjsm)
{
    const int bi = blockIdx.x;            // b*N + i
    const int b  = bi >> 7;
    const int lane = threadIdx.x;
    const float lm = lam[0];
    float v[4];
    float mx = -INFINITY;
    #pragma unroll
    for (int j4 = 0; j4 < 4; ++j4) {
        const int j = lane + 32 * j4;
        float a = (mask[b * N_ + j] == 0) ? -INFINITY
                                          : (-lm * adj[(size_t)bi * N_ + j]);
        v[j4] = a;
        mx = fmaxf(mx, a);
    }
    #pragma unroll
    for (int off = 16; off > 0; off >>= 1) mx = fmaxf(mx, __shfl_xor(mx, off, 32));
    float sum = 0.f;
    #pragma unroll
    for (int j4 = 0; j4 < 4; ++j4) { v[j4] = __expf(v[j4] - mx); sum += v[j4]; }
    #pragma unroll
    for (int off = 16; off > 0; off >>= 1) sum += __shfl_xor(sum, off, 32);
    const float inv = 1.0f / sum;
    #pragma unroll
    for (int j4 = 0; j4 < 4; ++j4)
        adjsm[(size_t)bi * N_ + lane + 32 * j4] = v[j4] * inv;
}

// ---------------------------------------------------------------------------
// TDM: 1-D bulk load of the 128 KB f16 WkT (global) into LDS.
// D# per CDNA5 ISA 8.3/8.4: group0 {count=1, lds_addr, global_addr, type=2},
// group1 {data_size=8B, tensor_dim0=tile_dim0=16384 units}.
// ---------------------------------------------------------------------------
__device__ inline void tdm_load_wk(const _Float16* src, _Float16* dstLds) {
#if HAVE_TDM
    const unsigned long long ga = (unsigned long long)(uintptr_t)src;
    const unsigned lds = (unsigned)(uintptr_t)dstLds;   // low 32 bits = LDS offset
    v4u g0;
    g0[0] = 1u;                                          // count=1, is_restore=0
    g0[1] = lds;                                         // lds_addr [63:32]
    g0[2] = (unsigned)(ga & 0xFFFFFFFFu);                // global_addr [95:64]
    g0[3] = (unsigned)((ga >> 32) & 0x1FFFFFFu)          // global_addr [120:96]
          | (2u << 30);                                  // type=2 ("image")
    v8i_ g1;
    g1[0] = 0x00030000;                  // workgroup_mask=0, data_size=3 (8B)
    g1[1] = (int)(16384u << 16);         // tensor_dim0[15:0] in [63:48]
    g1[2] = (int)(1u << 16);             // tensor_dim0 hi=0 | tensor_dim1=1
    g1[3] = (int)(16384u << 16);         // tile_dim0=16384 in [127:112]
    g1[4] = 0;                           // tile_dim1=0 (unused), tile_dim2=0
    g1[5] = 16384;                       // tensor_dim0_stride lo32
    g1[6] = 0;
    g1[7] = 0;
    v4i_ z4 = {0, 0, 0, 0};
#if defined(__clang_major__) && (__clang_major__ >= 23)
    v8i_ z8 = {0, 0, 0, 0, 0, 0, 0, 0};
    __builtin_amdgcn_tensor_load_to_lds(g0, g1, z4, z4, z8, 0);
#else
    __builtin_amdgcn_tensor_load_to_lds(g0, g1, z4, z4, 0);
#endif
#else
    (void)src; (void)dstLds;
#endif
}

// ---------------------------------------------------------------------------
// Kernel D (the big one): per (b,m) slab,
//   KE = f16( edges_att[b,m,:,:] ) @ f16(Wk)     (128x256 @ 256x256, WMMA f16)
//   out_scores[b,h,m,n] = qs[b,m,:] . (KE[n,:]+bk)
//   in_scores [b,h,n,m] = qs[b,n,:] . (KE[n,:]+bk)
// 256 threads (8 waves), 256 KB dynamic LDS:
//   WkT f16 [256][256] (128KB, TDM) | E f16 [128][256] (64KB) | KE f16 [d][n] (64KB)
// ---------------------------------------------------------------------------
__global__ __launch_bounds__(256) void gt_ke_scores_kernel(
    const float* __restrict__ edges, const _Float16* __restrict__ wkT,
    const float* __restrict__ qs, const float* __restrict__ bk,
    float* __restrict__ sout, float* __restrict__ sin_)
{
    extern __shared__ char smem[];
    _Float16* WK = (_Float16*)smem;           // 65536 halfs
    _Float16* E  = WK + 65536;                // 32768 halfs
    _Float16* KE = E  + 32768;                // 32768 halfs, layout [d][n]

    const int bm = blockIdx.x;                // b*N + m
    const int b  = bm >> 7;
    const int m  = bm & (N_ - 1);
    const int t  = threadIdx.x;

    // ---- wave 0 kicks the TDM bulk load of WkT into LDS ----
#if HAVE_TDM
    if (t < 32) {
        tdm_load_wk(wkT, WK);
    }
#else
    {
        const uint4* src = (const uint4*)wkT;     // 8192 uint4
        uint4* dst = (uint4*)WK;
        for (int i = t; i < 8192; i += 256) dst[i] = src[i];
    }
#endif

    // ---- all waves: stage E slab: fp32 global -> f16 LDS, row-major [row][k] ----
    {
        const float4* src = (const float4*)(edges + (size_t)bm * (N_ * D_)); // 8192 float4
        for (int i = t; i < 8192; i += 256) {
            float4 f = src[i];
            PK4 pk;
            pk.h[0] = (_Float16)f.x; pk.h[1] = (_Float16)f.y;
            pk.h[2] = (_Float16)f.z; pk.h[3] = (_Float16)f.w;
            *(uint2*)(E + i * 4) = pk.u;
        }
    }
#if HAVE_TDM
    if (t < 32) {
        __builtin_amdgcn_s_wait_tensorcnt(0);
    }
#endif
    __syncthreads();

    // ---- WMMA GEMM: wave w owns rows [16w, 16w+16) of KE ----
    const int wave = t >> 5;
    const int lane = t & 31;
    const int lm   = lane & 15;      // position within 16
    const int lh   = lane >> 4;      // half-wave select

    // A fragments: 16-bit A 16x32 layout: lane l -> M=l%16,
    // K in {kb..kb+7, kb+16..kb+23}, kb=(l/16)*8; register resident for all 8 K-steps.
    V16 afrag[8];
    {
        const _Float16* ebase = E + (wave * 16 + lm) * D_;
        #pragma unroll
        for (int k0 = 0; k0 < 8; ++k0) {
            const _Float16* p = ebase + k0 * 32 + lh * 8;
            afrag[k0].q[0] = *(const uint4*)(p);
            afrag[k0].q[1] = *(const uint4*)(p + 16);
        }
    }

    for (int nt = 0; nt < 16; ++nt) {
        v8f c = {};
        // B 32x16 layout: lane l -> N=l%16, K in {kb..kb+7, kb+16..kb+23}, kb=(l/16)*8 (from WkT[n][k])
        const _Float16* wb = WK + (nt * 16 + lm) * D_ + lh * 8;
        #pragma unroll
        for (int k0 = 0; k0 < 8; ++k0) {
            V16 bfrag;
            const _Float16* p = wb + k0 * 32;
            bfrag.q[0] = *(const uint4*)(p);
            bfrag.q[1] = *(const uint4*)(p + 16);
            c = __builtin_amdgcn_wmma_f32_16x16x32_f16(
                    false, afrag[k0].v, false, bfrag.v,
                    (short)0, c, false, false);
        }
        // C layout: VGPR r -> row = 16*wave + r + 8*lh, col = nt*16 + lm.
        // Store to KE column-major [d][n]: 8 consecutive rows -> one b128.
        PK8 pk;
        #pragma unroll
        for (int r = 0; r < 8; ++r) pk.h[r] = (_Float16)c[r];
        const int col = nt * 16 + lm;
        *(uint4*)(KE + col * N_ + wave * 16 + lh * 8) = pk.q;
    }
    __syncthreads();

    // ---- dot phase: thread t -> n = t%128, heads (t/128)*4 .. +3 ----
    const int n  = t & (N_ - 1);
    const int hg = t >> 7;
    const float* qm = qs + (size_t)(b * N_ + m) * D_;
    const float* qn = qs + (size_t)(b * N_ + n) * D_;
    for (int hh = 0; hh < 4; ++hh) {
        const int h = hg * 4 + hh;
        float ao = 0.f, ai = 0.f;
        #pragma unroll 8
        for (int dk = 0; dk < DK_; ++dk) {
            const int d = h * DK_ + dk;
            const float kev = (float)KE[d * N_ + n] + bk[d];
            ao = fmaf(kev, qm[d], ao);
            ai = fmaf(kev, qn[d], ai);
        }
        sout[((size_t)(b * H_ + h) * N_ + m) * N_ + n] = ao;
        sin_[((size_t)(b * H_ + h) * N_ + n) * N_ + m] = ai;
    }
}

// ---------------------------------------------------------------------------
// Kernel E: per (b,h,i) row: softmax both score rows, combine, scale by adjsm
// ---------------------------------------------------------------------------
__global__ __launch_bounds__(32) void gt_combine_kernel(
    const float* __restrict__ sout, const float* __restrict__ sin_,
    const float* __restrict__ adjsm, const int* __restrict__ mask,
    float* __restrict__ p)
{
    const int row = blockIdx.x;          // (b*H + h)*N + i
    const int i   = row & (N_ - 1);
    const int bh  = row >> 7;
    const int b   = bh >> 3;
    const int lane = threadIdx.x;
    const float* so = sout + (size_t)row * N_;
    const float* si = sin_ + (size_t)row * N_;

    float vo[4], vi[4];
    float mo = -INFINITY, mi = -INFINITY;
    #pragma unroll
    for (int j4 = 0; j4 < 4; ++j4) {
        const int j = lane + 32 * j4;
        const bool mk = mask[b * N_ + j] != 0;
        float a = mk ? so[j] : -INFINITY;
        float c = mk ? si[j] : -INFINITY;
        vo[j4] = a; vi[j4] = c;
        mo = fmaxf(mo, a); mi = fmaxf(mi, c);
    }
    #pragma unroll
    for (int off = 16; off > 0; off >>= 1) {
        mo = fmaxf(mo, __shfl_xor(mo, off, 32));
        mi = fmaxf(mi, __shfl_xor(mi, off, 32));
    }
    float sumo = 0.f, sumi = 0.f;
    #pragma unroll
    for (int j4 = 0; j4 < 4; ++j4) {
        vo[j4] = __expf(vo[j4] - mo); sumo += vo[j4];
        vi[j4] = __expf(vi[j4] - mi); sumi += vi[j4];
    }
    #pragma unroll
    for (int off = 16; off > 0; off >>= 1) {
        sumo += __shfl_xor(sumo, off, 32);
        sumi += __shfl_xor(sumi, off, 32);
    }
    const float invo = 1.0f / sumo, invi = 1.0f / sumi;
    #pragma unroll
    for (int j4 = 0; j4 < 4; ++j4) {
        const int j = lane + 32 * j4;
        const float oa = vo[j4] * invo;
        const float ia = vi[j4] * invi;
        float pv = oa + ia - ((j == i) ? oa : 0.f);
        p[(size_t)row * N_ + j] = pv * adjsm[(size_t)(b * N_ + i) * N_ + j];
    }
}

// ---------------------------------------------------------------------------
// Kernel F: y[b,m,d] = sum_n p[b, d/32, m, n] * vlin[b,n,d]
// ---------------------------------------------------------------------------
__global__ __launch_bounds__(256) void gt_pv_kernel(
    const float* __restrict__ p, const float* __restrict__ vlin,
    float* __restrict__ y)
{
    const int idx = blockIdx.x * 256 + threadIdx.x;   // B*N*D
    const int d  = idx & (D_ - 1);
    const int bm = idx >> 8;
    const int b  = bm >> 7;
    const int m  = bm & (N_ - 1);
    const int h  = d >> 5;
    const float* prow = p + ((size_t)(b * H_ + h) * N_ + m) * N_;
    const float* vcol = vlin + (size_t)b * N_ * D_ + d;
    float acc = 0.f;
    for (int nn = 0; nn < N_; ++nn)
        acc = fmaf(prow[nn], vcol[nn * D_], acc);
    y[idx] = acc;
}

// ---------------------------------------------------------------------------
// Kernel G: out = y @ Wo + bo
// ---------------------------------------------------------------------------
__global__ __launch_bounds__(256) void gt_out_kernel(
    const float* __restrict__ y, const float* __restrict__ Wo,
    const float* __restrict__ bo, float* __restrict__ out)
{
    const int idx = blockIdx.x * 256 + threadIdx.x;   // B*N*D
    const int d  = idx & (D_ - 1);
    const int bn = idx >> 8;
    const float* yrow = y + (size_t)bn * D_;
    float acc = bo[d];
    for (int k = 0; k < D_; ++k)
        acc = fmaf(yrow[k], Wo[k * D_ + d], acc);
    out[idx] = acc;
}

// ---------------------------------------------------------------------------
extern "C" void kernel_launch(void* const* d_in, const int* in_sizes, int n_in,
                              void* d_out, int out_size, void* d_ws, size_t ws_size,
                              hipStream_t stream) {
    (void)in_sizes; (void)n_in; (void)out_size; (void)ws_size;
    const float* query = (const float*)d_in[0];
    // d_in[1] = key (unused by reference)
    const float* value = (const float*)d_in[2];
    const float* adj   = (const float*)d_in[3];
    const float* edges = (const float*)d_in[4];
    const int*   mask  = (const int*)  d_in[5];
    const float* Wq    = (const float*)d_in[6];
    const float* bq    = (const float*)d_in[7];
    const float* Wk    = (const float*)d_in[8];
    const float* bk    = (const float*)d_in[9];
    const float* Wv    = (const float*)d_in[10];
    const float* bv    = (const float*)d_in[11];
    const float* Wo    = (const float*)d_in[12];
    const float* bo    = (const float*)d_in[13];
    const float* lam   = (const float*)d_in[14];
    float* out = (float*)d_out;

    // workspace layout (bytes)
    char* ws = (char*)d_ws;
    float*     qs    = (float*)(ws);                       //  2 MB  (B*N*D)
    float*     vlin  = (float*)(ws + (2u  << 20));         //  2 MB
    float*     adjsm = (float*)(ws + (4u  << 20));         //  1 MB  (B*N*N)
    _Float16*  wkT   = (_Float16*)(ws + (5u << 20));       // 128 KB (D*D f16)
    float*     sout  = (float*)(ws + (6u  << 20));         //  8 MB  (B*H*N*N)
    float*     sin_  = (float*)(ws + (14u << 20));         //  8 MB
    float*     pbuf  = (float*)(ws + (22u << 20));         //  8 MB
    float*     ybuf  = (float*)(ws + (30u << 20));         //  2 MB

    // A: projections (B*N*D threads)
    gt_qv_kernel<<<dim3((B_ * N_ * D_) / 256), dim3(256), 0, stream>>>(
        query, value, Wq, bq, Wv, bv, qs, vlin);

    // B: pack Wk -> f16 transposed
    gt_pack_wk_kernel<<<dim3((D_ * D_) / 256), dim3(256), 0, stream>>>(Wk, wkT);

    // C: adjacency softmax (one wave per (b,i))
    gt_adj_softmax_kernel<<<dim3(B_ * N_), dim3(32), 0, stream>>>(adj, mask, lam, adjsm);

    // D: big WMMA kernel, one WG per (b,m), 256 KB dynamic LDS
    gt_ke_scores_kernel<<<dim3(B_ * N_), dim3(256), 262144, stream>>>(
        edges, wkT, qs, bk, sout, sin_);

    // E: softmax + combine (one wave per (b,h,i) row)
    gt_combine_kernel<<<dim3(B_ * H_ * N_), dim3(32), 0, stream>>>(
        sout, sin_, adjsm, mask, pbuf);

    // F: p @ v
    gt_pv_kernel<<<dim3((B_ * N_ * D_) / 256), dim3(256), 0, stream>>>(pbuf, vlin, ybuf);

    // G: output projection
    gt_out_kernel<<<dim3((B_ * N_ * D_) / 256), dim3(256), 0, stream>>>(ybuf, Wo, bo, out);
}
